// Module_1_1151051235416
// MI455X (gfx1250) — matrix operand order repaired and checked by reading, not compiled, not run on hardware
//
#include <hip/hip_runtime.h>

typedef __attribute__((ext_vector_type(16))) __bf16 v16bf;
typedef __attribute__((ext_vector_type(8)))  float  v8f;

#define D 128

// ---------------------------------------------------------------------------
// prep: W (f32 [k][n]) -> Wt (bf16 [n][k]) for both layers; zero BN stats.
// grid 64 x 256 == 16384 threads == D*D
// ---------------------------------------------------------------------------
__global__ void prep_kernel(const float* __restrict__ W1,
                            const float* __restrict__ W2,
                            __bf16* __restrict__ Wt1,
                            __bf16* __restrict__ Wt2,
                            float* __restrict__ stats) {
  int i = blockIdx.x * blockDim.x + threadIdx.x;   // i = k*128 + n
  int k = i >> 7, n = i & 127;
  Wt1[n * D + k] = (__bf16)W1[i];
  Wt2[n * D + k] = (__bf16)W2[i];
  if (i < 1024) stats[i] = 0.f;                    // 8 x 128 stat slots
}

// ---------------------------------------------------------------------------
// agg = epsilon * v   (also initializes the scatter accumulator)
// ---------------------------------------------------------------------------
__global__ void init_agg_kernel(const float4* __restrict__ v4,
                                const float* __restrict__ eps,
                                float4* __restrict__ agg4, long total4) {
  float e = eps[0];
  long i = (long)blockIdx.x * blockDim.x + threadIdx.x;
  long stride = (long)gridDim.x * blockDim.x;
  for (; i < total4; i += stride) {
    float4 x = v4[i];
    agg4[i] = make_float4(e * x.x, e * x.y, e * x.z, e * x.w);
  }
}

// ---------------------------------------------------------------------------
// scatter-add: 32 lanes per edge, one float4 per lane, L2-resident atomics
// ---------------------------------------------------------------------------
__global__ void scatter_kernel(const float4* __restrict__ v4,
                               const float* __restrict__ ew,
                               const int* __restrict__ esrc,
                               const int* __restrict__ edst,
                               float* __restrict__ agg, int E) {
  long gid = (long)blockIdx.x * blockDim.x + threadIdx.x;
  int e = (int)(gid >> 5);
  if (e >= E) return;
  int lane = (int)(gid & 31);
  int src = esrc[e];
  int dst = edst[e];
  float w = ew[e];
  float4 x = v4[(long)src * 32 + lane];
  float* p = agg + (long)dst * D + lane * 4;
  __hip_atomic_fetch_add(p + 0, w * x.x, __ATOMIC_RELAXED, __HIP_MEMORY_SCOPE_AGENT);
  __hip_atomic_fetch_add(p + 1, w * x.y, __ATOMIC_RELAXED, __HIP_MEMORY_SCOPE_AGENT);
  __hip_atomic_fetch_add(p + 2, w * x.z, __ATOMIC_RELAXED, __HIP_MEMORY_SCOPE_AGENT);
  __hip_atomic_fetch_add(p + 3, w * x.w, __ATOMIC_RELAXED, __HIP_MEMORY_SCOPE_AGENT);
}

// ---------------------------------------------------------------------------
// Y[N,128] = transform(A[N,128]) @ W[128,128] + bias
// transform: identity (0) or relu(scale*x + shift) (1)
// Block = 256 thr = 8 waves; each wave owns one 16x16 output tile of a
// 16-row x 128-col block. K=128 swept with 4x v_wmma_f32_16x16x32_bf16.
// Results go through LDS for coalesced float4 global stores (+bias).
// In-place safe (Y==A): block reads only its own 16 rows, all staged to LDS
// before any store.
// ---------------------------------------------------------------------------
__global__ void __launch_bounds__(256)
gemm_kernel(const float* __restrict__ A, const __bf16* __restrict__ Wt,
            const float* __restrict__ bias, float* __restrict__ Y, int N,
            const float* __restrict__ tscale, const float* __restrict__ tshift,
            int transform) {
  __shared__ __align__(32) __bf16 As[16][136];   // padded: conflict-free b128 reads
  __shared__ __align__(16) float  Cs[16][136];   // padded: 16B-aligned rows
  int tid = threadIdx.x;
  int rowbase = blockIdx.x * 16;

  // ---- Stage A tile (f32 -> optional BN-ReLU -> bf16) into LDS.
  // Stride 256 over 16x128: column is fixed per thread -> hoist BN params.
  {
    int c = tid & 127;
    float tsc = 0.f, tsh = 0.f;
    if (transform) { tsc = tscale[c]; tsh = tshift[c]; }
    for (int r = tid >> 7; r < 16; r += 2) {
      int row = rowbase + r;
      float x = (row < N) ? A[(long)row * D + c] : 0.f;
      if (transform) x = fmaxf(fmaf(x, tsc, tsh), 0.f);
      As[r][c] = (__bf16)x;
    }
  }
  __syncthreads();

  int wave = tid >> 5, lane = tid & 31;
  int c0   = wave * 16;          // this wave's output-column tile
  int half = lane >> 4;          // lane group 0/1 per ISA fragment layout
  int rlo  = lane & 15;
  int kb8  = half * 8;

  v8f acc;
#pragma unroll
  for (int i = 0; i < 8; ++i) acc[i] = 0.f;

  const __bf16* wcol = Wt + (long)(c0 + rlo) * D;   // B col = lane&15, contiguous K

#pragma unroll
  for (int kc = 0; kc < 4; ++kc) {
    union { v16bf v; float4 f[2]; } a, b;
    // A 16x32 bf16 fragment: elem j -> K = kc*32 + half*16 + j, row = rlo
    a.f[0] = *(const float4*)&As[rlo][kc * 32 + half * 16];
    a.f[1] = *(const float4*)&As[rlo][kc * 32 + half * 16 + 8];
    // B 32x16 bf16 fragment: elem j -> K = kc*32 + half*16 + j, col = rlo
    int k0 = kc * 32 + half * 16;
    b.f[0] = *(const float4*)(wcol + k0);
    b.f[1] = *(const float4*)(wcol + k0 + 8);
    acc = __builtin_amdgcn_wmma_f32_16x16x32_bf16(false, a.v, false, b.v,
                                                  (short)0, acc, false, false);
  }

  // ---- Park the 16x16 tile in LDS (C/D layout: VGPR i -> row half*8 + i)
#pragma unroll
  for (int i = 0; i < 8; ++i) Cs[half * 8 + i][c0 + rlo] = acc[i];
  __syncthreads();

  // ---- Coalesced float4 stores with bias; single guard per row (none when
  //      the tile is full, which is always true for N % 16 == 0).
  const float4* bias4 = (const float4*)bias;
  bool full = (rowbase + 16 <= N);
  for (int idx = tid; idx < 16 * 32; idx += 256) {
    int r = idx >> 5, cq = idx & 31;
    int row = rowbase + r;
    if (full || row < N) {
      float4 x = *(const float4*)&Cs[r][cq * 4];
      float4 bb = bias4[cq];
      x.x += bb.x; x.y += bb.y; x.z += bb.z; x.w += bb.w;
      *(float4*)&Y[(long)row * D + cq * 4] = x;
    }
  }
}

// ---------------------------------------------------------------------------
// per-column sum / sumsq over N rows (for BatchNorm batch stats)
// ---------------------------------------------------------------------------
__global__ void colstats_kernel(const float* __restrict__ X, int N,
                                float* __restrict__ sum, float* __restrict__ sq) {
  __shared__ float ss[256], qs[256];
  int t = threadIdx.x;
  int col = t & 127, rg = t >> 7;
  float s = 0.f, q = 0.f;
  for (int row = blockIdx.x * 2 + rg; row < N; row += gridDim.x * 2) {
    float x = X[(long)row * D + col];
    s += x;
    q += x * x;
  }
  ss[t] = s; qs[t] = q;
  __syncthreads();
  if (t < 128) {
    s = ss[t] + ss[t + 128];
    q = qs[t] + qs[t + 128];
    __hip_atomic_fetch_add(&sum[col], s, __ATOMIC_RELAXED, __HIP_MEMORY_SCOPE_AGENT);
    __hip_atomic_fetch_add(&sq[col],  q, __ATOMIC_RELAXED, __HIP_MEMORY_SCOPE_AGENT);
  }
}

// fold mean/var/gamma/beta -> per-column scale & shift    (1 block, 128 thr)
__global__ void finalize_kernel(const float* __restrict__ sum,
                                const float* __restrict__ sq,
                                const float* __restrict__ gamma,
                                const float* __restrict__ beta,
                                float* __restrict__ scale,
                                float* __restrict__ shift, int N) {
  int c = threadIdx.x;
  float invN = 1.f / (float)N;
  float mean = sum[c] * invN;
  float var  = sq[c] * invN - mean * mean;
  float sc   = gamma[c] * rsqrtf(var + 1e-5f);
  scale[c] = sc;
  shift[c] = beta[c] - mean * sc;
}

// out = relu(scale*out + shift), in place, float4
__global__ void apply_kernel(float4* __restrict__ Y4,
                             const float4* __restrict__ scale4,
                             const float4* __restrict__ shift4, long total4) {
  long i = (long)blockIdx.x * blockDim.x + threadIdx.x;
  long stride = (long)gridDim.x * blockDim.x;
  for (; i < total4; i += stride) {
    int cq = (int)(i & 31);
    float4 x = Y4[i];
    float4 sc = scale4[cq];
    float4 sh = shift4[cq];
    x.x = fmaxf(fmaf(x.x, sc.x, sh.x), 0.f);
    x.y = fmaxf(fmaf(x.y, sc.y, sh.y), 0.f);
    x.z = fmaxf(fmaf(x.z, sc.z, sh.z), 0.f);
    x.w = fmaxf(fmaf(x.w, sc.w, sh.w), 0.f);
    Y4[i] = x;
  }
}

extern "C" void kernel_launch(void* const* d_in, const int* in_sizes, int n_in,
                              void* d_out, int out_size, void* d_ws, size_t ws_size,
                              hipStream_t stream) {
  const float* v    = (const float*)d_in[0];
  const float* ew   = (const float*)d_in[1];
  const float* eps  = (const float*)d_in[2];
  const float* W1   = (const float*)d_in[3];
  const float* b1   = (const float*)d_in[4];
  const float* g1   = (const float*)d_in[5];
  const float* be1  = (const float*)d_in[6];
  const float* W2   = (const float*)d_in[7];
  const float* b2   = (const float*)d_in[8];
  const float* g2   = (const float*)d_in[9];
  const float* be2  = (const float*)d_in[10];
  const int*   esrc = (const int*)d_in[11];
  const int*   edst = (const int*)d_in[12];
  float* out = (float*)d_out;

  int N = in_sizes[0] / D;
  int E = in_sizes[1];

  // workspace layout: agg[N*D] f32 | Wt1,Wt2 bf16[D*D] | stats f32[1024]
  char* ws = (char*)d_ws;
  float* agg = (float*)ws;
  size_t aggBytes = (((size_t)N * D * sizeof(float)) + 255) & ~(size_t)255;
  __bf16* Wt1 = (__bf16*)(ws + aggBytes);
  __bf16* Wt2 = Wt1 + D * D;
  float* stats = (float*)(ws + aggBytes + 2 * (size_t)D * D * sizeof(__bf16));
  float* sum1 = stats, *sq1 = stats + 128, *sc1 = stats + 256, *sh1 = stats + 384;
  float* sum2 = stats + 512, *sq2 = stats + 640, *sc2 = stats + 768, *sh2 = stats + 896;

  long total4 = (long)N * (D / 4);
  int ewBlocks = (int)((total4 + 255) / 256);
  int scBlocks = (int)(((long)E * 32 + 255) / 256);
  int gmBlocks = (N + 15) / 16;

  prep_kernel<<<64, 256, 0, stream>>>(W1, W2, Wt1, Wt2, stats);
  init_agg_kernel<<<ewBlocks, 256, 0, stream>>>((const float4*)v, eps,
                                                (float4*)agg, total4);
  scatter_kernel<<<scBlocks, 256, 0, stream>>>((const float4*)v, ew, esrc, edst,
                                               agg, E);
  // layer 1: agg = agg @ W1 + b1 (in place), then BN stats
  gemm_kernel<<<gmBlocks, 256, 0, stream>>>(agg, Wt1, b1, agg, N,
                                            nullptr, nullptr, 0);
  colstats_kernel<<<1024, 256, 0, stream>>>(agg, N, sum1, sq1);
  finalize_kernel<<<1, 128, 0, stream>>>(sum1, sq1, g1, be1, sc1, sh1, N);
  // layer 2: out = relu(bn(agg)) @ W2 + b2 (BN-ReLU fused into A staging)
  gemm_kernel<<<gmBlocks, 256, 0, stream>>>(agg, Wt2, b2, out, N, sc1, sh1, 1);
  colstats_kernel<<<1024, 256, 0, stream>>>(out, N, sum2, sq2);
  finalize_kernel<<<1, 128, 0, stream>>>(sum2, sq2, g2, be2, sc2, sh2, N);
  apply_kernel<<<ewBlocks, 256, 0, stream>>>((float4*)out, (const float4*)sc2,
                                             (const float4*)sh2, total4);
}
